// TransformerBlock_26285199852245
// MI455X (gfx1250) — matrix operand order repaired and checked by reading, not compiled, not run on hardware
//
#include <hip/hip_runtime.h>
#include <math.h>

// ---------------------------------------------------------------------------
// Transformer block (pre-LN, causal MHA, GLU-GELU FFN) for MI455X / gfx1250.
// Matmuls on v_wmma_f32_16x16x32_bf16; GEMM tiles staged by the Tensor Data
// Mover (tensor_load_to_lds, TENSORcnt) with double-buffered LDS.
// ---------------------------------------------------------------------------

#define Bc 4
#define Sc 2048
#define Dc 2048
#define Hc 16
#define DKc 128
#define DFFc 8192
#define Mrows (Bc * Sc)   // 8192 token rows
#define ThreeD (3 * Dc)   // 6144
#define TwoDFF (2 * DFFc) // 16384
#define LNEPS 1e-5f

typedef __bf16 bf16_t;
typedef bf16_t v16bf __attribute__((ext_vector_type(16)));
typedef float  v8f   __attribute__((ext_vector_type(8)));
typedef int    v4i   __attribute__((ext_vector_type(4)));
typedef unsigned int u32x4 __attribute__((ext_vector_type(4)));
typedef int    i32x8 __attribute__((ext_vector_type(8)));
typedef int    i32x4 __attribute__((ext_vector_type(4)));
typedef unsigned short u16;

// Fragment = 16 bf16 per lane = 8 VGPRs; overlay views for 16B LDS/global moves.
union Frag {
  v16bf v;
  v4i   q[2];
  u16   e[16];
};

__device__ __forceinline__ u16 f2bf(float f) {
  unsigned int u = __float_as_uint(f);
  u += 0x7FFFu + ((u >> 16) & 1u);   // round-to-nearest-even
  return (u16)(u >> 16);
}

// ---------------------------------------------------------------------------
// TDM: DMA one bf16 tile of 128 rows x 32 cols into LDS, with hardware row
// padding: +16B after every 64B => LDS row stride 80B (= 40 u16), matching
// the WMMA fragment addressing below.  D# layout per CDNA5 ISA ch.8.
// This toolchain's builtin takes 6 args (g0, g1, g2, g3, g4, cpol).
// ---------------------------------------------------------------------------
__device__ __forceinline__ void tdm_load_tile_128x32(const u16* gsrc,
                                                     unsigned lds_off,
                                                     int tensor_rows, int Kdim) {
  unsigned long long ga = (unsigned long long)(size_t)gsrc;  // tile start (bytes)
  u32x4 g0;
  g0[0] = 1u;                                    // count=1, user descriptor
  g0[1] = lds_off;                               // lds_addr
  g0[2] = (unsigned)ga;                          // global_addr[31:0]
  g0[3] = (unsigned)((ga >> 32) & 0x01FFFFFFull) | 0x80000000u; // addr[56:32] | type=2
  i32x8 g1;
  g1[0] = (1 << 16)            // data_size = 2 bytes
        | (1 << 20)            // pad_enable
        | (3 << 22)            // pad_interval: 16 DWORDs (64B) between pads
        | (3 << 25);           // pad_amount: 4 DWORDs (16B)
  g1[1] = (int)((unsigned)(Kdim & 0xFFFF) << 16);                  // tensor_dim0[15:0]
  g1[2] = (int)(((unsigned)(Kdim >> 16) & 0xFFFFu) |
                ((unsigned)(tensor_rows & 0xFFFF) << 16));         // dim0 hi | dim1 lo
  g1[3] = (int)(((unsigned)(tensor_rows >> 16) & 0xFFFFu) |
                (32u << 16));                                      // dim1 hi | tile_dim0=32
  g1[4] = 128;                                                     // tile_dim1=128, tile_dim2=0
  g1[5] = Kdim;                                                    // tensor_dim0_stride[31:0]
  g1[6] = 0;                                                       // stride hi | dim1_stride lo
  g1[7] = 0;
  i32x4 z4;
  z4[0] = 0; z4[1] = 0; z4[2] = 0; z4[3] = 0;
  i32x8 z8;
  z8[0] = 0; z8[1] = 0; z8[2] = 0; z8[3] = 0;
  z8[4] = 0; z8[5] = 0; z8[6] = 0; z8[7] = 0;
  __builtin_amdgcn_tensor_load_to_lds(g0, g1, z4, z4, z8, 0);
}

// ---------------------------------------------------------------------------
// f32 -> bf16 conversion (weights)
// ---------------------------------------------------------------------------
__global__ void cvt_f32_bf16(const float* __restrict__ in, u16* __restrict__ out, int n) {
  int i = blockIdx.x * blockDim.x + threadIdx.x;
  int stride = gridDim.x * blockDim.x;
  for (; i < n; i += stride) out[i] = f2bf(in[i]);
}

// ---------------------------------------------------------------------------
// LayerNorm (one row of D=2048 per 256-thread block) -> bf16
// ---------------------------------------------------------------------------
__global__ __launch_bounds__(256) void ln_bf16(const float* __restrict__ x,
                                               const float* __restrict__ g,
                                               const float* __restrict__ bvec,
                                               u16* __restrict__ y) {
  const int row = blockIdx.x;
  const int t = threadIdx.x;
  const float* xr = x + (size_t)row * Dc;
  float vals[8];
  float s = 0.f, ss = 0.f;
#pragma unroll
  for (int i = 0; i < 8; ++i) {
    float v = xr[t + i * 256];
    vals[i] = v; s += v; ss += v * v;
  }
#pragma unroll
  for (int off = 16; off > 0; off >>= 1) {
    s  += __shfl_down(s, off, 32);
    ss += __shfl_down(ss, off, 32);
  }
  __shared__ float red[16];
  __shared__ float stats[2];
  int wave = t >> 5, lane = t & 31;
  if (lane == 0) { red[wave] = s; red[8 + wave] = ss; }
  __syncthreads();
  if (t == 0) {
    float a = 0.f, b2 = 0.f;
    for (int w = 0; w < 8; ++w) { a += red[w]; b2 += red[8 + w]; }
    float mean = a / (float)Dc;
    float var = b2 / (float)Dc - mean * mean;
    stats[0] = mean; stats[1] = rsqrtf(var + LNEPS);
  }
  __syncthreads();
  const float mean = stats[0], rstd = stats[1];
  u16* yr = y + (size_t)row * Dc;
#pragma unroll
  for (int i = 0; i < 8; ++i) {
    int c = t + i * 256;
    yr[c] = f2bf((vals[i] - mean) * rstd * g[c] + bvec[c]);
  }
}

// ---------------------------------------------------------------------------
// GEMM: C[M,N] = A[M,K] * W[N,K]^T  (+bias, +residual), bf16 in, WMMA f32 acc.
// Block tile 128x128, K-step 32, 256 threads = 8 waves, wave tile 32x64.
// Tiles DMA'd by TDM into double-buffered LDS; wave 0 drives the DMA queue.
// ---------------------------------------------------------------------------
__global__ __launch_bounds__(256) void gemm_bf16_wmma(
    const u16* __restrict__ A, const u16* __restrict__ W,
    const float* __restrict__ bias, const float* __restrict__ residual,
    float* __restrict__ outF, u16* __restrict__ outB,
    int M, int N, int K) {
  __shared__ __align__(16) u16 lA[2][128 * 40];  // 80B row stride (TDM padded)
  __shared__ __align__(16) u16 lB[2][128 * 40];

  const int t = threadIdx.x;
  const int lane = t & 31;
  const int wave = t >> 5;
  const int waveM = wave & 3;        // 4 waves in M
  const int waveN = wave >> 2;       // 2 waves in N
  const int m0 = blockIdx.y * 128, n0 = blockIdx.x * 128;

  v8f acc[2][4];
#pragma unroll
  for (int i = 0; i < 2; ++i)
#pragma unroll
    for (int j = 0; j < 4; ++j)
#pragma unroll
      for (int r = 0; r < 8; ++r) acc[i][j][r] = 0.f;

  const int l16 = lane & 15;
  const int h8 = (lane >> 4) * 8;        // A-fragment K split by lane half
  const int koffB = (lane >> 4) * 8;     // B-fragment K split by lane half
  const int rowA0 = waveM * 32 + l16;
  const int rowB0 = waveN * 64 + l16;

  const unsigned ldsA0 = (unsigned)(size_t)(void*)&lA[0][0];
  const unsigned ldsA1 = (unsigned)(size_t)(void*)&lA[1][0];
  const unsigned ldsB0 = (unsigned)(size_t)(void*)&lB[0][0];
  const unsigned ldsB1 = (unsigned)(size_t)(void*)&lB[1][0];

  const int nk = K >> 5;
  if (wave == 0) {
    tdm_load_tile_128x32(&A[(size_t)m0 * K], ldsA0, M, K);
    tdm_load_tile_128x32(&W[(size_t)n0 * K], ldsB0, N, K);
  }

  for (int it = 0; it < nk; ++it) {
    const int buf = it & 1;
    if (wave == 0) __builtin_amdgcn_s_wait_tensorcnt(0);
    __syncthreads();                     // tiles[buf] ready; prev compute done
    if (wave == 0 && it + 1 < nk) {      // prefetch next K tile into buf^1
      int k1 = (it + 1) << 5;
      tdm_load_tile_128x32(&A[(size_t)m0 * K + k1], buf ? ldsA0 : ldsA1, M, K);
      tdm_load_tile_128x32(&W[(size_t)n0 * K + k1], buf ? ldsB0 : ldsB1, N, K);
    }

    const u16* la = &lA[buf][0];
    const u16* lb = &lB[buf][0];
    Frag a[2], bfr[4];
#pragma unroll
    for (int mi = 0; mi < 2; ++mi) {
      int r = rowA0 + mi * 16;
      a[mi].q[0] = *(const v4i*)&la[r * 40 + h8];
      a[mi].q[1] = *(const v4i*)&la[r * 40 + 16 + h8];
    }
#pragma unroll
    for (int ni = 0; ni < 4; ++ni) {
      int r = rowB0 + ni * 16;
      bfr[ni].q[0] = *(const v4i*)&lb[r * 40 + koffB];
      bfr[ni].q[1] = *(const v4i*)&lb[r * 40 + 16 + koffB];
    }
#pragma unroll
    for (int mi = 0; mi < 2; ++mi)
#pragma unroll
      for (int ni = 0; ni < 4; ++ni)
        acc[mi][ni] = __builtin_amdgcn_wmma_f32_16x16x32_bf16(
            false, a[mi].v, false, bfr[ni].v, (short)0, acc[mi][ni], false, false);
    __syncthreads();                     // done reading buf before next DMA hits it
  }

  // Epilogue: C/D layout -> lane holds column n=lane%16, VGPR r -> row r + half*8
#pragma unroll
  for (int mi = 0; mi < 2; ++mi)
#pragma unroll
    for (int ni = 0; ni < 4; ++ni) {
      int n = n0 + waveN * 64 + ni * 16 + l16;
      float bv = bias ? bias[n] : 0.f;
#pragma unroll
      for (int r = 0; r < 8; ++r) {
        int m = m0 + waveM * 32 + mi * 16 + r + (lane >> 4) * 8;
        float v = acc[mi][ni][r] + bv;
        if (residual) v += residual[(size_t)m * N + n];
        if (outF) outF[(size_t)m * N + n] = v;
        else      outB[(size_t)m * N + n] = f2bf(v);
      }
    }
}

// ---------------------------------------------------------------------------
// Flash attention (causal). Block = 128 threads = 4 waves, 16 q rows / wave,
// 64 q rows / block. Computes S^T = K*Q^T (lane = query column), online
// softmax with per-lane stats, O^T = V^T * P accumulated via WMMA.
// ---------------------------------------------------------------------------
__global__ __launch_bounds__(128) void flash_attn(const u16* __restrict__ qkv,
                                                  u16* __restrict__ out) {
  __shared__ __align__(16) u16 vt[128 * 40];  // V^T tile: [dk][key], padded rows

  const int t = threadIdx.x;
  const int lane = t & 31, wave = t >> 5;
  const int qt = blockIdx.x, h = blockIdx.y, b = blockIdx.z;
  const int q0b = qt * 64;
  const int q0 = q0b + wave * 16;
  const int l16 = lane & 15, hh = lane >> 4, h8 = hh * 8;
  const float scale = 0.08838834764831845f;  // 1/sqrt(128)

  // Q as 4 B-fragments (lane = query col, dk h8..h8+7 and 16+h8..16+h8+7 per half), in regs.
  Frag qf[4];
  {
    const u16* qrow = qkv + ((size_t)(b * Sc + q0 + l16)) * ThreeD + h * DKc;
#pragma unroll
    for (int c = 0; c < 4; ++c) {
      qf[c].q[0] = *(const v4i*)&qrow[c * 32 + h8];
      qf[c].q[1] = *(const v4i*)&qrow[c * 32 + 16 + h8];
    }
  }

  v8f o[8];
#pragma unroll
  for (int i = 0; i < 8; ++i)
#pragma unroll
    for (int r = 0; r < 8; ++r) o[i][r] = 0.f;
  float mrun = -3.0e38f, lsum = 0.f;
  const int qg = q0 + l16;

  const int kend = q0b + 64;
  for (int kt = 0; kt < kend; kt += 32) {
    // Stage V^T into LDS (32 keys x 128 dk, transposed).
    {
      int kk = t & 31, ch = t >> 5;  // ch in 0..3 -> dk chunk of 32
      const u16* vrow = qkv + ((size_t)(b * Sc + kt + kk)) * ThreeD + 2 * Dc + h * DKc + ch * 32;
#pragma unroll
      for (int j = 0; j < 4; ++j) {
        Frag u; u.q[0] = *(const v4i*)&vrow[j * 8];
#pragma unroll
        for (int i = 0; i < 8; ++i) vt[(ch * 32 + j * 8 + i) * 40 + kk] = u.e[i];
      }
    }
    __syncthreads();

    // S^T = K * Q^T for two 16-key sub-tiles (accumulated over dk=128).
    v8f s0, s1;
#pragma unroll
    for (int r = 0; r < 8; ++r) { s0[r] = 0.f; s1[r] = 0.f; }
    {
      const u16* krow0 = qkv + ((size_t)(b * Sc + kt + l16)) * ThreeD + Dc + h * DKc;
      const u16* krow1 = krow0 + (size_t)16 * ThreeD;
#pragma unroll
      for (int c = 0; c < 4; ++c) {
        Frag ka;
        ka.q[0] = *(const v4i*)&krow0[c * 32 + h8];
        ka.q[1] = *(const v4i*)&krow0[c * 32 + 16 + h8];
        s0 = __builtin_amdgcn_wmma_f32_16x16x32_bf16(false, ka.v, false, qf[c].v, (short)0, s0, false, false);
        ka.q[0] = *(const v4i*)&krow1[c * 32 + h8];
        ka.q[1] = *(const v4i*)&krow1[c * 32 + 16 + h8];
        s1 = __builtin_amdgcn_wmma_f32_16x16x32_bf16(false, ka.v, false, qf[c].v, (short)0, s1, false, false);
      }
    }

    // Scale + causal mask; per-lane stats (lane = query), one cross-half shfl.
    float pa[8], pb[8];
    float mt = -3.0e38f;
#pragma unroll
    for (int r = 0; r < 8; ++r) {
      int k0g = kt + r + h8;
      int k1g = k0g + 16;
      float v0 = (k0g <= qg) ? s0[r] * scale : -3.0e38f;
      float v1 = (k1g <= qg) ? s1[r] * scale : -3.0e38f;
      pa[r] = v0; pb[r] = v1;
      mt = fmaxf(mt, fmaxf(v0, v1));
    }
    mt = fmaxf(mt, __shfl_xor(mt, 16, 32));
    float mnew = fmaxf(mrun, mt);
    float corr = __expf(mrun - mnew);
    float tsum = 0.f;
#pragma unroll
    for (int r = 0; r < 8; ++r) {
      pa[r] = (pa[r] > -1.0e38f) ? __expf(pa[r] - mnew) : 0.f;
      pb[r] = (pb[r] > -1.0e38f) ? __expf(pb[r] - mnew) : 0.f;
      tsum += pa[r] + pb[r];
    }
    tsum += __shfl_xor(tsum, 16, 32);
    lsum = lsum * corr + tsum;
    mrun = mnew;

    // P (32 keys x 16 q) as B-fragment: lane = q col, K covered by lane half.
    Frag pf;
    {
      float pv[16];
#pragma unroll
      for (int r = 0; r < 8; ++r) {
        float qa = __shfl_xor(pa[r], 16, 32);
        float qb = __shfl_xor(pb[r], 16, 32);
        if (hh == 0) { pv[r] = pa[r]; pv[r + 8] = qa; }
        else         { pv[r] = qb;    pv[r + 8] = pb[r]; }
      }
#pragma unroll
      for (int i = 0; i < 16; ++i) pf.e[i] = f2bf(pv[i]);
    }

    // O^T = corr*O^T + V^T * P (8 dk-tiles of 16).
#pragma unroll
    for (int tt = 0; tt < 8; ++tt) {
#pragma unroll
      for (int r = 0; r < 8; ++r) o[tt][r] *= corr;
      Frag va;
      int dkr = tt * 16 + l16;
      va.q[0] = *(const v4i*)&vt[dkr * 40 + hh * 16];
      va.q[1] = *(const v4i*)&vt[dkr * 40 + hh * 16 + 8];
      o[tt] = __builtin_amdgcn_wmma_f32_16x16x32_bf16(false, va.v, false, pf.v, (short)0, o[tt], false, false);
    }
    __syncthreads();
  }

  // Normalize and write: O^T layout -> lane = q col, VGPR r -> dk row.
  const float inv = 1.f / lsum;
  u16* orow = out + ((size_t)(b * Sc + qg)) * Dc + h * DKc;
#pragma unroll
  for (int tt = 0; tt < 8; ++tt)
#pragma unroll
    for (int r = 0; r < 8; ++r) {
      int dk = tt * 16 + r + h8;
      orow[dk] = f2bf(o[tt][r] * inv);
    }
}

// ---------------------------------------------------------------------------
// GLU: f = gelu(a) * gelu(g), erf-GELU, bf16 out.
// ---------------------------------------------------------------------------
__global__ void glu_gelu(const float* __restrict__ u, u16* __restrict__ f, long long total) {
  long long i = (long long)blockIdx.x * blockDim.x + threadIdx.x;
  long long stride = (long long)gridDim.x * blockDim.x;
  for (; i < total; i += stride) {
    long long m = i / DFFc, j = i % DFFc;
    float a = u[m * TwoDFF + j];
    float g = u[m * TwoDFF + DFFc + j];
    float ga = 0.5f * a * (1.f + erff(a * 0.70710678118f));
    float gg = 0.5f * g * (1.f + erff(g * 0.70710678118f));
    f[i] = f2bf(ga * gg);
  }
}

// ---------------------------------------------------------------------------
// Orchestration
// ---------------------------------------------------------------------------
extern "C" void kernel_launch(void* const* d_in, const int* in_sizes, int n_in,
                              void* d_out, int out_size, void* d_ws, size_t ws_size,
                              hipStream_t stream) {
  (void)in_sizes; (void)n_in; (void)out_size; (void)ws_size;
  const float* x    = (const float*)d_in[0];
  // d_in[1] = mask (causal, hardcoded)
  const float* ln1g = (const float*)d_in[2];
  const float* ln1b = (const float*)d_in[3];
  const float* qkvw = (const float*)d_in[4];
  const float* wow  = (const float*)d_in[5];
  const float* ln2g = (const float*)d_in[6];
  const float* ln2b = (const float*)d_in[7];
  const float* w1w  = (const float*)d_in[8];
  const float* w1b  = (const float*)d_in[9];
  const float* w2w  = (const float*)d_in[10];
  const float* w2b  = (const float*)d_in[11];
  float* out = (float*)d_out;

  char* p = (char*)d_ws;
  auto alloc = [&](size_t bytes) -> char* {
    char* r = p; p += (bytes + 255) & ~(size_t)255; return r;
  };
  u16* qkvw_b = (u16*)alloc((size_t)ThreeD * Dc * 2);
  u16* wow_b  = (u16*)alloc((size_t)Dc * Dc * 2);
  u16* w1w_b  = (u16*)alloc((size_t)TwoDFF * Dc * 2);
  u16* w2w_b  = (u16*)alloc((size_t)Dc * DFFc * 2);
  u16* h_b    = (u16*)alloc((size_t)Mrows * Dc * 2);
  u16* qkv_b  = (u16*)alloc((size_t)Mrows * ThreeD * 2);
  u16* attn_b = (u16*)alloc((size_t)Mrows * Dc * 2);
  float* x1   = (float*)alloc((size_t)Mrows * Dc * 4);
  float* u_f  = (float*)alloc((size_t)Mrows * TwoDFF * 4);
  u16* f_b    = (u16*)alloc((size_t)Mrows * DFFc * 2);

  cvt_f32_bf16<<<2048, 256, 0, stream>>>(qkvw, qkvw_b, ThreeD * Dc);
  cvt_f32_bf16<<<2048, 256, 0, stream>>>(wow,  wow_b,  Dc * Dc);
  cvt_f32_bf16<<<4096, 256, 0, stream>>>(w1w,  w1w_b,  TwoDFF * Dc);
  cvt_f32_bf16<<<4096, 256, 0, stream>>>(w2w,  w2w_b,  Dc * DFFc);

  // ---- attention sublayer ----
  ln_bf16<<<Mrows, 256, 0, stream>>>(x, ln1g, ln1b, h_b);
  gemm_bf16_wmma<<<dim3(ThreeD / 128, Mrows / 128), 256, 0, stream>>>(
      h_b, qkvw_b, nullptr, nullptr, nullptr, qkv_b, Mrows, ThreeD, Dc);
  flash_attn<<<dim3(Sc / 64, Hc, Bc), 128, 0, stream>>>(qkv_b, attn_b);
  gemm_bf16_wmma<<<dim3(Dc / 128, Mrows / 128), 256, 0, stream>>>(
      attn_b, wow_b, nullptr, x, x1, nullptr, Mrows, Dc, Dc);

  // ---- GLU feed-forward sublayer ----
  ln_bf16<<<Mrows, 256, 0, stream>>>(x1, ln2g, ln2b, h_b);
  gemm_bf16_wmma<<<dim3(TwoDFF / 128, Mrows / 128), 256, 0, stream>>>(
      h_b, w1w_b, w1b, nullptr, u_f, nullptr, Mrows, TwoDFF, Dc);
  glu_gelu<<<4096, 256, 0, stream>>>(u_f, f_b, (long long)Mrows * DFFc);
  gemm_bf16_wmma<<<dim3(Dc / 128, Mrows / 128), 256, 0, stream>>>(
      f_b, w2w_b, w2b, x1, out, nullptr, Mrows, Dc, DFFc);
}
